// GEN_77584289235500
// MI455X (gfx1250) — matrix operand order repaired and checked by reading, not compiled, not run on hardware
//
#include <hip/hip_runtime.h>
#include <hip/hip_bf16.h>

// ---------------- types ----------------
typedef __attribute__((ext_vector_type(16))) __bf16 v16bf;
typedef __attribute__((ext_vector_type(8)))  __bf16 v8bf;
typedef __attribute__((ext_vector_type(8)))  float  v8f;
typedef __attribute__((ext_vector_type(4)))  float  v4f;

#define DEVINL __device__ __forceinline__

static constexpr int Bc  = 8;
static constexpr int Tc  = 8192;
static constexpr int Sc  = 128;
static constexpr int VDc = 256;
static constexpr int LDc = 768;
static constexpr int Ec  = 256;
static constexpr int Hc  = 8;
static constexpr int HDc = 32;
static constexpr float SCALEc = 0.17677669529663687f;  // 1/sqrt(32)
static constexpr float NEGc   = -9.0e15f;

DEVINL v8f v8f_zero() {
  v8f z;
#pragma unroll
  for (int i = 0; i < 8; ++i) z[i] = 0.0f;
  return z;
}

DEVINL v8f wmma_bf16(v16bf a, v16bf b, v8f c) {
  return __builtin_amdgcn_wmma_f32_16x16x32_bf16(false, a, false, b, (short)0, c,
                                                 false, false);
}

DEVINL v16bf cat8(v8bf lo, v8bf hi) {
  return __builtin_shufflevector(lo, hi, 0, 1, 2, 3, 4, 5, 6, 7, 8, 9, 10, 11,
                                 12, 13, 14, 15);
}

// A-operand per-lane layout: elems 0..7 = K[half*8 .. +7], 8..15 = K[16+half*8 .. +7]
DEVINL v16bf load_a16(const __bf16* base, int half) {
  v8bf lo = *(const v8bf*)(base + half * 8);
  v8bf hi = *(const v8bf*)(base + 16 + half * 8);
  return cat8(lo, hi);
}
// B-operand per-lane layout: elems 0..7 = K[half*8 .. +7], 8..15 = K[16+half*8 .. +7]
DEVINL v16bf load_b16(const __bf16* base, int half) {
  v8bf lo = *(const v8bf*)(base + half * 8);
  v8bf hi = *(const v8bf*)(base + 16 + half * 8);
  return cat8(lo, hi);
}

// =====================================================================
// Weight f32 -> bf16 pre-conversion (one pass; weights then live in L2)
// =====================================================================
__global__ __launch_bounds__(256) void cvt_f32_bf16(const float* __restrict__ in,
                                                    __bf16* __restrict__ out, int n) {
  const int i4 = (blockIdx.x * 256 + threadIdx.x) * 4;
  if (i4 < n) {
    v4f x = *(const v4f*)(in + i4);
#pragma unroll
    for (int i = 0; i < 4; ++i) out[i4 + i] = (__bf16)x[i];
  }
}

// =====================================================================
// WMMA GEMM:  Out[m,n] = ((A (+A2))[m,:] . W[n,:] + bias[n]) * scale
// A: [M,K] (f32 or bf16), W: [N,K] bf16 row-major, Out: [M,N] (bf16 or f32)
// One wave owns a 16-row M strip x 256-column N chunk (16 accumulators):
// A is read exactly once from HBM; W streams from L2.
// =====================================================================
template <bool A_IS_F32, bool FUSE_ADD, bool OUT_F32>
__global__ __launch_bounds__(128) void wmma_gemm(const void* __restrict__ Aptr,
                                                 const void* __restrict__ A2ptr,
                                                 const __bf16* __restrict__ W,
                                                 const float* __restrict__ bias,
                                                 void* __restrict__ Outp,
                                                 int M, int N, int K, float scale) {
  const int lane = threadIdx.x & 31;
  const int wid  = threadIdx.x >> 5;
  const int gw   = blockIdx.x * (blockDim.x >> 5) + wid;
  const int nChunks = N >> 8;  // 256-wide column chunks
  const int tm = gw / nChunks;
  const int nc = gw - tm * nChunks;
  if (tm >= (M >> 4)) return;
  const int r    = lane & 15;
  const int half = lane >> 4;
  const int arow = tm * 16 + r;

  v8f cacc[16];
#pragma unroll
  for (int nt = 0; nt < 16; ++nt) cacc[nt] = v8f_zero();

  for (int k0 = 0; k0 < K; k0 += 32) {
    v16bf a;
    if constexpr (A_IS_F32) {
      const float* Af = (const float*)Aptr + (size_t)arow * K + k0;
      v4f x0 = *(const v4f*)(Af + half * 8);
      v4f x1 = *(const v4f*)(Af + half * 8 + 4);
      v4f y0 = *(const v4f*)(Af + 16 + half * 8);
      v4f y1 = *(const v4f*)(Af + 16 + half * 8 + 4);
      if constexpr (FUSE_ADD) {
        const float* A2f = (const float*)A2ptr + (size_t)arow * K + k0;
        x0 += *(const v4f*)(A2f + half * 8);
        x1 += *(const v4f*)(A2f + half * 8 + 4);
        y0 += *(const v4f*)(A2f + 16 + half * 8);
        y1 += *(const v4f*)(A2f + 16 + half * 8 + 4);
      }
#pragma unroll
      for (int i = 0; i < 4; ++i) {
        a[i]      = (__bf16)x0[i];
        a[4 + i]  = (__bf16)x1[i];
        a[8 + i]  = (__bf16)y0[i];
        a[12 + i] = (__bf16)y1[i];
      }
    } else {
      const __bf16* Ab = (const __bf16*)Aptr + (size_t)arow * K + k0;
      a = load_a16(Ab, half);
    }
#pragma unroll
    for (int nt = 0; nt < 16; ++nt) {
      const int wrow = nc * 256 + nt * 16 + r;
      v16bf bm = load_b16(W + (size_t)wrow * K + k0, half);
      cacc[nt] = wmma_bf16(a, bm, cacc[nt]);
    }
  }

#pragma unroll
  for (int nt = 0; nt < 16; ++nt) {
    const int n  = nc * 256 + nt * 16 + r;
    const float bn = bias[n];
#pragma unroll
    for (int rr = 0; rr < 8; ++rr) {
      const int m = tm * 16 + half * 8 + rr;
      const float val = (cacc[nt][rr] + bn) * scale;
      if constexpr (OUT_F32)
        ((float*)Outp)[(size_t)m * N + n] = val;
      else
        ((__bf16*)Outp)[(size_t)m * N + n] = (__bf16)val;
    }
  }
}

// =====================================================================
// Fused bi-directional attention. One block per (b,h); 8 waves.
// Streams T in tiles of 128 rows (16 t-rows per wave).
//   scores = q @ k^T                            (WMMA)
//   out_v  = softmax_s(scores+addl) @ val_l     (row softmax in regs + WMMA)
//   out_l  = softmax_t(scores^T+addv) @ val_v   (online/flash over T + WMMA)
// =====================================================================
struct SLDS {
  __bf16 ksh[Sc][HDc];      //  8 KB  k[s][d]
  __bf16 vlT[HDc][Sc];      //  8 KB  val_l^T[d][s]
  __bf16 vvT[HDc][128];     //  8 KB  val_v tile^T[d][t_local]
  __bf16 pbuf[128][128];    // 32 KB  shared: P_v[t][s] then P_l[s][t]
  float  wstat[8][Sc];      //  4 KB  per-wave column stats
  float  m_s[Sc];
  float  d_s[Sc];
  float  scale_s[Sc];
  float  addl_sh[Sc];
  float  addv_sh[128];
};

__global__ __launch_bounds__(256) void biattn_kernel(
    const __bf16* __restrict__ q, const __bf16* __restrict__ ksrc,
    const __bf16* __restrict__ vv, const __bf16* __restrict__ vl,
    const int* __restrict__ mask_l, const int* __restrict__ mask_v,
    __bf16* __restrict__ ovh, __bf16* __restrict__ olh) {
  __shared__ SLDS s;

  const int blk = blockIdx.x;
  const int b = blk >> 3, h = blk & 7;
  const int tid = threadIdx.x;
  const int lane = tid & 31, w = tid >> 5;
  const int r = lane & 15, half = lane >> 4;

  // ---- stage k (vectorized) and val_l^T (coalesced load, transposed store) ----
  for (int i = tid; i < (Sc * HDc) / 8; i += 256) {   // 512 16B chunks
    const int ss = i >> 2, dc = (i & 3) * 8;
    const size_t g = ((size_t)(b * Sc + ss)) * Ec + h * HDc + dc;
    v8bf x = *(const v8bf*)&ksrc[g];
    *(v8bf*)&s.ksh[ss][dc] = x;
    v8bf y = *(const v8bf*)&vl[g];
#pragma unroll
    for (int e = 0; e < 8; ++e) s.vlT[dc + e][ss] = y[e];
  }
  if (tid < Sc) {
    const int mv = mask_l[b * Sc + tid];
    s.addl_sh[tid] = (mv == 0) ? NEGc : (float)mv;
    s.m_s[tid] = -3.0e38f;
    s.d_s[tid] = 0.0f;
  }
  __syncthreads();

  float al[8];
#pragma unroll
  for (int j = 0; j < 8; ++j) al[j] = s.addl_sh[j * 16 + r];

  v8f acc[2];
  acc[0] = v8f_zero();
  acc[1] = v8f_zero();

  for (int t0 = 0; t0 < Tc; t0 += 128) {
    // ---- stage val_v tile: coalesced v8bf global loads, transposed LDS store ----
    for (int i = tid; i < (128 * HDc) / 8; i += 256) {  // 512 16B chunks
      const int tt = i >> 2, dc = (i & 3) * 8;
      v8bf x = *(const v8bf*)&vv[((size_t)(b * Tc + t0 + tt)) * Ec + h * HDc + dc];
#pragma unroll
      for (int e = 0; e < 8; ++e) s.vvT[dc + e][tt] = x[e];
    }
    if (tid < 128) {
      const int mvv = mask_v[b * Tc + t0 + tid];
      s.addv_sh[tid] = (mvv == 0) ? NEGc : (float)mvv;
    }
    if (t0 + 128 < Tc)  // global_prefetch_b8 for next tile
      __builtin_prefetch(&vv[((size_t)(b * Tc + t0 + 128 + (tid & 127))) * Ec + h * HDc], 0, 1);
    __syncthreads();

    // ---- scores: 16 t-rows per wave vs all 128 s ----
    const __bf16* qp = q + ((size_t)(b * Tc + t0 + w * 16 + r)) * Ec + h * HDc;
    v16bf aq = load_a16(qp, half);
    v8f c[8];
#pragma unroll
    for (int j = 0; j < 8; ++j) {
      v16bf bk = load_b16(&s.ksh[j * 16 + r][0], half);
      c[j] = wmma_bf16(aq, bk, v8f_zero());
    }

    float av[8];
#pragma unroll
    for (int rr = 0; rr < 8; ++rr) av[rr] = s.addv_sh[w * 16 + half * 8 + rr];

    // ================= out_v: row softmax over s =================
    float rmax[8], rsum[8];
#pragma unroll
    for (int rr = 0; rr < 8; ++rr) rmax[rr] = -3.0e38f;
#pragma unroll
    for (int j = 0; j < 8; ++j)
#pragma unroll
      for (int rr = 0; rr < 8; ++rr) rmax[rr] = fmaxf(rmax[rr], c[j][rr] + al[j]);
#pragma unroll
    for (int m = 1; m <= 8; m <<= 1)
#pragma unroll
      for (int rr = 0; rr < 8; ++rr) rmax[rr] = fmaxf(rmax[rr], __shfl_xor(rmax[rr], m, 32));

#pragma unroll
    for (int rr = 0; rr < 8; ++rr) rsum[rr] = 0.0f;
#pragma unroll
    for (int j = 0; j < 8; ++j)
#pragma unroll
      for (int rr = 0; rr < 8; ++rr) {
        const float pe = __expf(c[j][rr] + al[j] - rmax[rr]);
        rsum[rr] += pe;
        s.pbuf[w * 16 + half * 8 + rr][j * 16 + r] = (__bf16)pe;  // P_v[t][s]
      }
#pragma unroll
    for (int m = 1; m <= 8; m <<= 1)
#pragma unroll
      for (int rr = 0; rr < 8; ++rr) rsum[rr] += __shfl_xor(rsum[rr], m, 32);
    __syncthreads();

    v8f ov[2];
    ov[0] = v8f_zero();
    ov[1] = v8f_zero();
#pragma unroll
    for (int nt = 0; nt < 2; ++nt)
#pragma unroll
      for (int kc = 0; kc < 4; ++kc) {
        v16bf ap = load_a16(&s.pbuf[w * 16 + r][kc * 32], half);
        v16bf bp = load_b16(&s.vlT[nt * 16 + r][kc * 32], half);
        ov[nt] = wmma_bf16(ap, bp, ov[nt]);
      }
#pragma unroll
    for (int nt = 0; nt < 2; ++nt)
#pragma unroll
      for (int rr = 0; rr < 8; ++rr) {
        const int tt = t0 + w * 16 + half * 8 + rr;
        ovh[((size_t)(b * Tc + tt)) * Ec + h * HDc + nt * 16 + r] =
            (__bf16)(ov[nt][rr] / rsum[rr]);
      }
    __syncthreads();  // pbuf reused as P_l next

    // ================= out_l: online softmax over t =================
    float cmax[8];
#pragma unroll
    for (int j = 0; j < 8; ++j) {
      float mx = -3.0e38f;
#pragma unroll
      for (int rr = 0; rr < 8; ++rr) mx = fmaxf(mx, c[j][rr] + av[rr]);
      mx = fmaxf(mx, __shfl_xor(mx, 16, 32));
      cmax[j] = mx;
    }
    if (half == 0)
#pragma unroll
      for (int j = 0; j < 8; ++j) s.wstat[w][j * 16 + r] = cmax[j];
    __syncthreads();
    if (tid < Sc) {
      float tm_ = s.wstat[0][tid];
#pragma unroll
      for (int ww = 1; ww < 8; ++ww) tm_ = fmaxf(tm_, s.wstat[ww][tid]);
      const float mo = s.m_s[tid];
      const float mn = fmaxf(mo, tm_);
      s.scale_s[tid] = __expf(mo - mn);
      s.m_s[tid] = mn;
    }
    __syncthreads();

    float csum[8];
#pragma unroll
    for (int j = 0; j < 8; ++j) {
      const float mn = s.m_s[j * 16 + r];
      v8bf pk;
      float sum = 0.0f;
#pragma unroll
      for (int rr = 0; rr < 8; ++rr) {
        const float pe = __expf(c[j][rr] + av[rr] - mn);
        sum += pe;
        pk[rr] = (__bf16)pe;
      }
      *(v8bf*)&s.pbuf[j * 16 + r][w * 16 + half * 8] = pk;  // P_l[s][t]
      csum[j] = sum;
    }
#pragma unroll
    for (int j = 0; j < 8; ++j) csum[j] += __shfl_xor(csum[j], 16, 32);
    if (half == 0)
#pragma unroll
      for (int j = 0; j < 8; ++j) s.wstat[w][j * 16 + r] = csum[j];
    __syncthreads();
    if (tid < Sc) {
      float ssum = 0.0f;
#pragma unroll
      for (int ww = 0; ww < 8; ++ww) ssum += s.wstat[ww][tid];
      s.d_s[tid] = s.d_s[tid] * s.scale_s[tid] + ssum;
    }
    __syncthreads();

#pragma unroll
    for (int rr = 0; rr < 8; ++rr) {
      const float sc = s.scale_s[w * 16 + half * 8 + rr];
      acc[0][rr] *= sc;
      acc[1][rr] *= sc;
    }
#pragma unroll
    for (int nt = 0; nt < 2; ++nt)
#pragma unroll
      for (int kc = 0; kc < 4; ++kc) {
        v16bf ap = load_a16(&s.pbuf[w * 16 + r][kc * 32], half);
        v16bf bp = load_b16(&s.vvT[nt * 16 + r][kc * 32], half);
        acc[nt] = wmma_bf16(ap, bp, acc[nt]);
      }
    __syncthreads();  // protect vvT/addv/pbuf for next tile
  }

  // ---- epilogue: normalize and write out_l heads ----
#pragma unroll
  for (int nt = 0; nt < 2; ++nt)
#pragma unroll
    for (int rr = 0; rr < 8; ++rr) {
      const int sl = w * 16 + half * 8 + rr;
      const float dd = s.d_s[sl];
      olh[((size_t)(b * Sc + sl)) * Ec + h * HDc + nt * 16 + r] =
          (__bf16)(acc[nt][rr] / dd);
    }
}

// =====================================================================
extern "C" void kernel_launch(void* const* d_in, const int* in_sizes, int n_in,
                              void* d_out, int out_size, void* d_ws, size_t ws_size,
                              hipStream_t stream) {
  (void)in_sizes; (void)n_in; (void)out_size; (void)ws_size;
  const float* v      = (const float*)d_in[0];
  const float* l      = (const float*)d_in[1];
  const float* v_pos  = (const float*)d_in[2];
  const int*   mask_l = (const int*)d_in[3];
  const int*   mask_v = (const int*)d_in[4];
  const float* Wv  = (const float*)d_in[5];
  const float* bv  = (const float*)d_in[6];
  const float* Wl  = (const float*)d_in[7];
  const float* bl  = (const float*)d_in[8];
  const float* Wvv = (const float*)d_in[9];
  const float* bvv = (const float*)d_in[10];
  const float* Wvl = (const float*)d_in[11];
  const float* bvl = (const float*)d_in[12];
  const float* Wov = (const float*)d_in[13];
  const float* bov = (const float*)d_in[14];
  const float* Wol = (const float*)d_in[15];
  const float* bol = (const float*)d_in[16];

  const int MT = Bc * Tc;   // 65536
  const int MS = Bc * Sc;   // 1024

  char* ws = (char*)d_ws;
  size_t off = 0;
  __bf16* qb  = (__bf16*)(ws + off); off += (size_t)MT * Ec * 2;  // 32 MB
  __bf16* vvb = (__bf16*)(ws + off); off += (size_t)MT * Ec * 2;  // 32 MB
  __bf16* kb  = (__bf16*)(ws + off); off += (size_t)MS * Ec * 2;  // 0.5 MB
  __bf16* vlb = (__bf16*)(ws + off); off += (size_t)MS * Ec * 2;  // 0.5 MB
  __bf16* ovh = (__bf16*)(ws + off); off += (size_t)MT * Ec * 2;  // 32 MB
  __bf16* olh = (__bf16*)(ws + off); off += (size_t)MS * Ec * 2;  // 0.5 MB
  // bf16 weights
  __bf16* Wv_b  = (__bf16*)(ws + off); off += (size_t)Ec * VDc * 2;
  __bf16* Wl_b  = (__bf16*)(ws + off); off += (size_t)Ec * LDc * 2;
  __bf16* Wvv_b = (__bf16*)(ws + off); off += (size_t)Ec * VDc * 2;
  __bf16* Wvl_b = (__bf16*)(ws + off); off += (size_t)Ec * LDc * 2;
  __bf16* Wov_b = (__bf16*)(ws + off); off += (size_t)VDc * Ec * 2;
  __bf16* Wol_b = (__bf16*)(ws + off); off += (size_t)LDc * Ec * 2;

  float* out_v = (float*)d_out;
  float* out_l = (float*)d_out + (size_t)MT * VDc;

  // ---- weight conversion (tiny; weights then L2-resident) ----
  cvt_f32_bf16<<<(Ec * VDc) / 1024, 256, 0, stream>>>(Wv,  Wv_b,  Ec * VDc);
  cvt_f32_bf16<<<(Ec * LDc) / 1024, 256, 0, stream>>>(Wl,  Wl_b,  Ec * LDc);
  cvt_f32_bf16<<<(Ec * VDc) / 1024, 256, 0, stream>>>(Wvv, Wvv_b, Ec * VDc);
  cvt_f32_bf16<<<(Ec * LDc) / 1024, 256, 0, stream>>>(Wvl, Wvl_b, Ec * LDc);
  cvt_f32_bf16<<<(VDc * Ec) / 1024, 256, 0, stream>>>(Wov, Wov_b, VDc * Ec);
  cvt_f32_bf16<<<(LDc * Ec) / 1024, 256, 0, stream>>>(Wol, Wol_b, LDc * Ec);

  // ---- input projections (bf16 intermediates). waves = (M/16)*(N/256) ----
  wmma_gemm<true, true, false><<<(MT / 16) / 4, 128, 0, stream>>>(
      v, v_pos, Wv_b, bv, qb, MT, Ec, VDc, SCALEc);            // q
  wmma_gemm<true, false, false><<<(MT / 16) / 4, 128, 0, stream>>>(
      v, nullptr, Wvv_b, bvv, vvb, MT, Ec, VDc, 1.0f);         // val_v
  wmma_gemm<true, false, false><<<(MS / 16) / 4, 128, 0, stream>>>(
      l, nullptr, Wl_b, bl, kb, MS, Ec, LDc, 1.0f);            // k
  wmma_gemm<true, false, false><<<(MS / 16) / 4, 128, 0, stream>>>(
      l, nullptr, Wvl_b, bvl, vlb, MS, Ec, LDc, 1.0f);         // val_l

  // ---- fused bidirectional attention ----
  biattn_kernel<<<Bc * Hc, 256, 0, stream>>>(qb, kb, vvb, vlb, mask_l, mask_v,
                                             ovh, olh);

  // ---- output projections (f32 -> d_out) ----
  wmma_gemm<false, false, true><<<(MT / 16) / 4, 128, 0, stream>>>(
      ovh, nullptr, Wov_b, bov, out_v, MT, VDc, Ec, 1.0f);
  wmma_gemm<false, false, true><<<((MS / 16) * 3) / 4, 128, 0, stream>>>(
      olh, nullptr, Wol_b, bol, out_l, MS, LDc, Ec, 1.0f);
}
